// Seq2Seq_66005057405448
// MI455X (gfx1250) — matrix operand order repaired and checked by reading, not compiled, not run on hardware
//
#include <hip/hip_runtime.h>

// ---------------------------------------------------------------------------
// Seq2Seq (GRU encoder + Bahdanau attention + GRU decoder + vocab projection)
// CDNA5 / gfx1250, wave32, WMMA f32_16x16x32_f16.
// ---------------------------------------------------------------------------

typedef __attribute__((ext_vector_type(16))) _Float16 v16h;
typedef __attribute__((ext_vector_type(8)))  _Float16 v8h;
typedef __attribute__((ext_vector_type(8)))  float    v8f;

#define V_   32000
#define E_   256
#define H_   512
#define B_   16
#define TE_  512
#define TD_  128
#define G3_  1536   // 3*H

// ---------------------------------------------------------------------------
// helpers
// ---------------------------------------------------------------------------
static __device__ __forceinline__ float sigmoid_f(float x) {
  return 1.0f / (1.0f + __expf(-x));
}
static __device__ __forceinline__ float tanh_f(float x) {
  x = fminf(fmaxf(x, -15.0f), 15.0f);
  float e = __expf(2.0f * x);
  return (e - 1.0f) / (e + 1.0f);
}

static __device__ __forceinline__ v8f wmma_f16(v16h a, v16h b, v8f c) {
  // D = A(16x32 f16) * B(32x16 f16) + C(16x16 f32)
  return __builtin_amdgcn_wmma_f32_16x16x32_f16(
      /*neg_a=*/false, a, /*neg_b=*/false, b,
      /*c_mod=*/(short)0, c, /*reuse_a=*/false, /*reuse_b=*/false);
}

// A fragment, 16x32 f16, row-major source A[M][lda].
// Lane l<16 : row=l,   K = {k0..k0+7, k0+16..k0+23}
// Lane l>=16: row=l-16,K = {k0+8..k0+15, k0+24..k0+31}
static __device__ __forceinline__ v16h frag_a(const _Float16* A, int lda, int k0) {
  const int lane = threadIdx.x & 31;
  const int row  = lane & 15;
  const int kb   = k0 + ((lane >> 4) << 3);
  const _Float16* p = A + (long)row * lda + kb;
  v8h lo = *(const v8h*)(p);        // K = kb .. kb+7
  v8h hi = *(const v8h*)(p + 16);   // K = kb+16 .. kb+23
  v16h r;
#pragma unroll
  for (int i = 0; i < 8; ++i) { r[i] = lo[i]; r[i + 8] = hi[i]; }
  return r;
}

// B fragment, 32x16 f16, from TRANSPOSED weights Bt[N][ldb] (row n = output col).
// Lane l<16 : col=n0+l,    K = {k0..k0+7, k0+16..k0+23}
// Lane l>=16: col=n0+l-16, K = {k0+8..k0+15, k0+24..k0+31}
static __device__ __forceinline__ v16h frag_bt(const _Float16* Bt, int ldb, int n0, int k0) {
  const int lane = threadIdx.x & 31;
  const int col  = n0 + (lane & 15);
  const int kb   = k0 + ((lane >> 4) << 3);
  const _Float16* p = Bt + (long)col * ldb + kb;
  v8h lo = *(const v8h*)(p);
  v8h hi = *(const v8h*)(p + 16);
  v16h r;
#pragma unroll
  for (int i = 0; i < 8; ++i) { r[i] = lo[i]; r[i + 8] = hi[i]; }
  return r;
}

// ---------------------------------------------------------------------------
// weight transpose + convert:  in[K][N] f32  ->  out[N][K] f16
// ---------------------------------------------------------------------------
__global__ void f32_to_f16_T(const float* __restrict__ in, _Float16* __restrict__ out,
                             int K, int N) {
  long idx = (long)blockIdx.x * blockDim.x + threadIdx.x;
  long total = (long)K * N;
  if (idx >= total) return;
  int k = (int)(idx / N);
  int n = (int)(idx % N);
  out[(long)n * K + k] = (_Float16)in[idx];
}

// embedding gather -> f16   out[row][e] = emb[ids[row]][e]
__global__ void gather_f16(const int* __restrict__ ids, const float* __restrict__ emb,
                           _Float16* __restrict__ out, long total, int E) {
  long idx = (long)blockIdx.x * blockDim.x + threadIdx.x;
  if (idx >= total) return;
  long r = idx / E;
  int  e = (int)(idx % E);
  out[idx] = (_Float16)emb[(long)ids[r] * E + e];
}

// ---------------------------------------------------------------------------
// Generic WMMA GEMM:  C[M][N] = A[M][K](f16) * Bt[N][ldb](f16, transposed) + bias
// Block: 256 threads = 8 waves as 2(M) x 4(N); block tile 32 x 256.
// Each wave computes 16x64. Software pipelined: fragments for k+1 are issued
// before the four wmma of step k, so wmma waits only on already-landed loads.
// Requires M % 32 == 0, N % 256 == 0, K % 32 == 0.
// ---------------------------------------------------------------------------
__global__ __launch_bounds__(256) void gemm_wmma(
    const _Float16* __restrict__ A, const _Float16* __restrict__ Bt,
    const float* __restrict__ bias, float* __restrict__ C,
    _Float16* __restrict__ C16, int M, int N, int K, int ldb) {
  const int wave = threadIdx.x >> 5;
  const int lane = threadIdx.x & 31;
  const int wm = wave >> 2, wn = wave & 3;
  const long m0 = (long)blockIdx.y * 32 + wm * 16;
  const long n0 = (long)blockIdx.x * 256 + (long)wn * 64;
  const _Float16* Arow = A + m0 * K;

  v8f acc[4] = {};

  // prologue: fragments for k0 = 0
  v16h a0 = frag_a(Arow, K, 0);
  v16h b0 = frag_bt(Bt, ldb, (int)(n0 +  0), 0);
  v16h b1 = frag_bt(Bt, ldb, (int)(n0 + 16), 0);
  v16h b2 = frag_bt(Bt, ldb, (int)(n0 + 32), 0);
  v16h b3 = frag_bt(Bt, ldb, (int)(n0 + 48), 0);

  for (int k0 = 32; k0 < K; k0 += 32) {
    // issue next-step loads first
    v16h an = frag_a(Arow, K, k0);
    v16h c0 = frag_bt(Bt, ldb, (int)(n0 +  0), k0);
    v16h c1 = frag_bt(Bt, ldb, (int)(n0 + 16), k0);
    v16h c2 = frag_bt(Bt, ldb, (int)(n0 + 32), k0);
    v16h c3 = frag_bt(Bt, ldb, (int)(n0 + 48), k0);
    // compute current step
    acc[0] = wmma_f16(a0, b0, acc[0]);
    acc[1] = wmma_f16(a0, b1, acc[1]);
    acc[2] = wmma_f16(a0, b2, acc[2]);
    acc[3] = wmma_f16(a0, b3, acc[3]);
    a0 = an; b0 = c0; b1 = c1; b2 = c2; b3 = c3;
  }
  // epilogue
  acc[0] = wmma_f16(a0, b0, acc[0]);
  acc[1] = wmma_f16(a0, b1, acc[1]);
  acc[2] = wmma_f16(a0, b2, acc[2]);
  acc[3] = wmma_f16(a0, b3, acc[3]);

  const int cn = lane & 15;
  const int rbase = (lane < 16) ? 0 : 8;
#pragma unroll
  for (int t = 0; t < 4; ++t) {
    long colg = n0 + t * 16 + cn;
    float bs = bias ? bias[colg] : 0.0f;
#pragma unroll
    for (int v = 0; v < 8; ++v) {
      long o = (m0 + rbase + v) * (long)N + colg;
      float val = acc[t][v] + bs;
      if (C)   C[o]   = val;
      if (C16) C16[o] = (_Float16)val;
    }
  }
}

// ---------------------------------------------------------------------------
// Encoder scan: 1 persistent workgroup, 1024 threads = 32 waves.
// Wave w owns h-columns [16w,16w+16) and computes the z/r/h gate tiles for
// that j-range of gh = h @ enc_rk, so GRU gating stays in registers.
// gx (= x@enc_k + bi, all steps) is precomputed in global memory.
// ---------------------------------------------------------------------------
__global__ __launch_bounds__(1024) void enc_scan(
    const float* __restrict__ gx,        // [B*TE, 1536]
    const _Float16* __restrict__ rkT,    // [1536, 512] transposed f16
    const float* __restrict__ br,        // [1536]
    const int* __restrict__ ids,         // [B*TE]
    _Float16* __restrict__ out16,        // [B*TE, 512]
    float* __restrict__ h_final)         // [B*H]
{
  __shared__ float    hs[B_][H_];
  __shared__ _Float16 hh[B_][H_];
  const int tid = threadIdx.x;
  for (int i = tid; i < B_ * H_; i += 1024) {
    hs[i >> 9][i & 511] = 0.0f;
    hh[i >> 9][i & 511] = (_Float16)0.0f;
  }
  __syncthreads();

  const int wave = tid >> 5, lane = tid & 31;
  const int j0   = wave * 16;
  const int col  = j0 + (lane & 15);
  const int rbase = (lane < 16) ? 0 : 8;
  const float brz = br[col], brr = br[H_ + col], brh = br[2 * H_ + col];

  for (int t = 0; t < TE_; ++t) {
    v8f az = {}, ar = {}, ah = {};
#pragma unroll
    for (int kk = 0; kk < H_ / 32; ++kk) {
      const int k0 = kk * 32;
      v16h a = frag_a(&hh[0][0], H_, k0);
      az = wmma_f16(a, frag_bt(rkT, H_, j0,            k0), az);
      ar = wmma_f16(a, frag_bt(rkT, H_, H_ + j0,       k0), ar);
      ah = wmma_f16(a, frag_bt(rkT, H_, 2 * H_ + j0,   k0), ah);
    }
    float hn[8];
#pragma unroll
    for (int v = 0; v < 8; ++v) {
      const int b = rbase + v;
      const long g = ((long)b * TE_ + t) * G3_;
      float xz = gx[g + col];
      float xr = gx[g + H_ + col];
      float xh = gx[g + 2 * H_ + col];
      float ho = hs[b][col];
      float z = sigmoid_f(xz + az[v] + brz);
      float r = sigmoid_f(xr + ar[v] + brr);
      float c = tanh_f(xh + r * (ah[v] + brh));
      float hv = z * ho + (1.0f - z) * c;
      hn[v] = (ids[b * TE_ + t] != 0) ? hv : ho;
    }
    __syncthreads();   // all waves done reading hh for this step
#pragma unroll
    for (int v = 0; v < 8; ++v) {
      const int b = rbase + v;
      hs[b][col] = hn[v];
      _Float16 hf = (_Float16)hn[v];
      hh[b][col] = hf;
      out16[((long)b * TE_ + t) * H_ + col] = hf;
    }
    __syncthreads();
  }
  for (int i = tid; i < B_ * H_; i += 1024) h_final[i] = hs[i >> 9][i & 511];
}

// ---------------------------------------------------------------------------
// Decoder scan: 1 persistent workgroup, 1024 threads = 32 waves.
// Per step: q=h@W1 (WMMA) -> scores tanh(q+keys)@Va (VALU, v8h loads) ->
// softmax -> context (VALU, v8h loads) -> gi=ctx@dec_k[:H], gh=h@dec_rk
// (WMMA) -> gates in registers. gxe (embed part of gin@dec_k + bi) precomputed.
// ---------------------------------------------------------------------------
__global__ __launch_bounds__(1024) void dec_scan(
    const float* __restrict__ gxe,       // [B*TD, 1536]
    const _Float16* __restrict__ dkT,    // [1536, 768] transposed f16
    const _Float16* __restrict__ rkT,    // [1536, 512]
    const float* __restrict__ br,        // dec_br [1536]
    const _Float16* __restrict__ W1T,    // [512, 512]
    const float* __restrict__ b1,        // [512]
    const _Float16* __restrict__ keys16, // [B*TE, 512]
    const _Float16* __restrict__ enc16,  // [B*TE, 512]
    const float* __restrict__ Va,        // [512]
    const float* __restrict__ bV,        // [1]
    const int* __restrict__ ids,         // [B*TD]
    const float* __restrict__ h0,        // [B*H]
    _Float16* __restrict__ out16)        // [B*TD, 512]
{
  __shared__ float    hs[B_][H_];       // 32 KB
  __shared__ _Float16 hh[B_][H_];       // 16 KB
  __shared__ _Float16 qc[B_][H_];       // 16 KB (q, then context)
  __shared__ float    sc[B_][TE_];      // 32 KB (scores -> attn)
  __shared__ float    vas[H_];          // 2 KB (Va staged once)

  const int tid = threadIdx.x;
  for (int i = tid; i < B_ * H_; i += 1024) {
    float v0 = h0[i];
    hs[i >> 9][i & 511] = v0;
    hh[i >> 9][i & 511] = (_Float16)v0;
  }
  for (int i = tid; i < H_; i += 1024) vas[i] = Va[i];
  __syncthreads();

  const int wave = tid >> 5, lane = tid & 31;
  const int j0   = wave * 16;
  const int col  = j0 + (lane & 15);
  const int rbase = (lane < 16) ? 0 : 8;
  const float bVs = bV[0];
  const float brz = br[col], brr = br[H_ + col], brh = br[2 * H_ + col];
  const float b1c = b1[col];
  const int brow = tid >> 6;            // fixed batch row per thread (16 x 64)
  const int span = (tid & 63) * 8;      // 8-wide contiguous span in [0,512)

  for (int t = 0; t < TD_; ++t) {
    // ---- Phase 1: q = h @ W1 + b1 -> qc (f16)
    {
      v8f aq = {};
#pragma unroll
      for (int kk = 0; kk < H_ / 32; ++kk) {
        v16h a = frag_a(&hh[0][0], H_, kk * 32);
        aq = wmma_f16(a, frag_bt(W1T, H_, j0, kk * 32), aq);
      }
#pragma unroll
      for (int v = 0; v < 8; ++v) qc[rbase + v][col] = (_Float16)(aq[v] + b1c);
    }
    __syncthreads();

    // ---- Phase 2: scores[b][te] = tanh(q[b]+keys[b,te]) . Va + bV
    // thread owns batch row `brow`, te in [span, span+8)
#pragma unroll
    for (int j = 0; j < 8; ++j) {
      const int te = span + j;
      const _Float16* kp = keys16 + ((long)brow * TE_ + te) * H_;
      float acc = 0.0f;
      for (int x = 0; x < H_; x += 8) {
        v8h kv = *(const v8h*)(kp + x);
        v8h qv = *(const v8h*)(&qc[brow][x]);
#pragma unroll
        for (int i = 0; i < 8; ++i)
          acc += tanh_f((float)qv[i] + (float)kv[i]) * vas[x + i];
      }
      sc[brow][te] = acc + bVs;
    }
    __syncthreads();

    // ---- Phase 3: softmax over te, one wave per batch row
    if (wave < B_) {
      const int b = wave;
      float vals[16];
      float mx = -3.4e38f;
#pragma unroll
      for (int i = 0; i < 16; ++i) { vals[i] = sc[b][lane + i * 32]; mx = fmaxf(mx, vals[i]); }
#pragma unroll
      for (int off = 16; off > 0; off >>= 1) mx = fmaxf(mx, __shfl_xor(mx, off, 32));
      float sum = 0.0f;
#pragma unroll
      for (int i = 0; i < 16; ++i) { vals[i] = __expf(vals[i] - mx); sum += vals[i]; }
#pragma unroll
      for (int off = 16; off > 0; off >>= 1) sum += __shfl_xor(sum, off, 32);
      const float inv = 1.0f / sum;
#pragma unroll
      for (int i = 0; i < 16; ++i) sc[b][lane + i * 32] = vals[i] * inv;
    }
    __syncthreads();

    // ---- Phase 4: context[b][h] = sum_te attn * enc_out -> qc (reused)
    // thread owns batch row `brow`, h in [span, span+8): one b128 per te step
    {
      float acc[8] = {};
      const _Float16* ep = enc16 + (long)brow * TE_ * H_ + span;
      for (int te = 0; te < TE_; ++te) {
        const float w = sc[brow][te];
        v8h ev = *(const v8h*)(ep + (long)te * H_);
#pragma unroll
        for (int i = 0; i < 8; ++i) acc[i] += w * (float)ev[i];
      }
#pragma unroll
      for (int i = 0; i < 8; ++i) qc[brow][span + i] = (_Float16)acc[i];
    }
    __syncthreads();

    // ---- Phase 5: gi = ctx @ dec_k[:H], gh = h @ dec_rk, gates in registers
    v8f iz = {}, ir = {}, ih = {}, gz = {}, gr = {}, gh = {};
#pragma unroll
    for (int kk = 0; kk < H_ / 32; ++kk) {
      const int k0 = kk * 32;
      v16h a = frag_a(&qc[0][0], H_, k0);
      iz = wmma_f16(a, frag_bt(dkT, 768, j0,          k0), iz);
      ir = wmma_f16(a, frag_bt(dkT, 768, H_ + j0,     k0), ir);
      ih = wmma_f16(a, frag_bt(dkT, 768, 2 * H_ + j0, k0), ih);
    }
#pragma unroll
    for (int kk = 0; kk < H_ / 32; ++kk) {
      const int k0 = kk * 32;
      v16h a = frag_a(&hh[0][0], H_, k0);
      gz = wmma_f16(a, frag_bt(rkT, H_, j0,          k0), gz);
      gr = wmma_f16(a, frag_bt(rkT, H_, H_ + j0,     k0), gr);
      gh = wmma_f16(a, frag_bt(rkT, H_, 2 * H_ + j0, k0), gh);
    }
    float hn[8], on[8];
#pragma unroll
    for (int v = 0; v < 8; ++v) {
      const int b = rbase + v;
      const long g = ((long)b * TD_ + t) * G3_;
      float xz = iz[v] + gxe[g + col];              // gxe includes dec_bi
      float xr = ir[v] + gxe[g + H_ + col];
      float xh = ih[v] + gxe[g + 2 * H_ + col];
      float ho = hs[b][col];
      float z = sigmoid_f(xz + gz[v] + brz);
      float r = sigmoid_f(xr + gr[v] + brr);
      float c = tanh_f(xh + r * (gh[v] + brh));
      float hv = z * ho + (1.0f - z) * c;
      const bool m = ids[b * TD_ + t] != 0;
      hn[v] = m ? hv : ho;
      on[v] = m ? hv : 0.0f;
    }
    __syncthreads();
#pragma unroll
    for (int v = 0; v < 8; ++v) {
      const int b = rbase + v;
      hs[b][col] = hn[v];
      hh[b][col] = (_Float16)hn[v];
      out16[((long)b * TD_ + t) * H_ + col] = (_Float16)on[v];
    }
    __syncthreads();
  }
}

// ---------------------------------------------------------------------------
// launch
// ---------------------------------------------------------------------------
extern "C" void kernel_launch(void* const* d_in, const int* in_sizes, int n_in,
                              void* d_out, int out_size, void* d_ws, size_t ws_size,
                              hipStream_t stream) {
  (void)in_sizes; (void)n_in; (void)out_size; (void)ws_size;

  const int*   enc_ids = (const int*)d_in[0];
  const int*   dec_ids = (const int*)d_in[1];
  const float* emb     = (const float*)d_in[2];
  const float* enc_k   = (const float*)d_in[3];
  const float* enc_rk  = (const float*)d_in[4];
  const float* enc_bi  = (const float*)d_in[5];
  const float* enc_br  = (const float*)d_in[6];
  const float* dec_k   = (const float*)d_in[7];
  const float* dec_rk  = (const float*)d_in[8];
  const float* dec_bi  = (const float*)d_in[9];
  const float* dec_br  = (const float*)d_in[10];
  const float* W1      = (const float*)d_in[11];
  const float* b1      = (const float*)d_in[12];
  const float* W2      = (const float*)d_in[13];
  const float* b2      = (const float*)d_in[14];
  const float* Va      = (const float*)d_in[15];
  const float* bV      = (const float*)d_in[16];
  const float* Wd      = (const float*)d_in[17];
  const float* bd      = (const float*)d_in[18];
  float* out = (float*)d_out;

  char* p = (char*)d_ws;
  auto take = [&](size_t bytes) -> void* {
    void* r = (void*)p;
    p += (bytes + 255) & ~(size_t)255;
    return r;
  };

  _Float16* enc_emb16 = (_Float16*)take((size_t)B_ * TE_ * E_ * 2);
  _Float16* dec_emb16 = (_Float16*)take((size_t)B_ * TD_ * E_ * 2);
  _Float16* enc_kT    = (_Float16*)take((size_t)G3_ * E_ * 2);
  _Float16* enc_rkT   = (_Float16*)take((size_t)G3_ * H_ * 2);
  _Float16* dec_kT    = (_Float16*)take((size_t)G3_ * (H_ + E_) * 2);
  _Float16* dec_rkT   = (_Float16*)take((size_t)G3_ * H_ * 2);
  _Float16* W1T       = (_Float16*)take((size_t)H_ * H_ * 2);
  _Float16* W2T       = (_Float16*)take((size_t)H_ * H_ * 2);
  _Float16* WdT       = (_Float16*)take((size_t)V_ * H_ * 2);
  float*    gx_enc    = (float*)take((size_t)B_ * TE_ * G3_ * 4);
  float*    gxe       = (float*)take((size_t)B_ * TD_ * G3_ * 4);
  _Float16* enc16     = (_Float16*)take((size_t)B_ * TE_ * H_ * 2);
  _Float16* keys16    = (_Float16*)take((size_t)B_ * TE_ * H_ * 2);
  _Float16* decout16  = (_Float16*)take((size_t)B_ * TD_ * H_ * 2);
  float*    h_final   = (float*)take((size_t)B_ * H_ * 4);

  const dim3 blk(256);

  // weight transpose+convert to f16 [N][K]
  auto T = [&](const float* in, _Float16* o, int K, int N) {
    long tot = (long)K * N;
    f32_to_f16_T<<<dim3((unsigned)((tot + 255) / 256)), blk, 0, stream>>>(in, o, K, N);
  };
  T(enc_k,  enc_kT,  E_,       G3_);
  T(enc_rk, enc_rkT, H_,       G3_);
  T(dec_k,  dec_kT,  H_ + E_,  G3_);
  T(dec_rk, dec_rkT, H_,       G3_);
  T(W1,     W1T,     H_,       H_);
  T(W2,     W2T,     H_,       H_);
  T(Wd,     WdT,     H_,       V_);

  // embedding gathers
  {
    long tot = (long)B_ * TE_ * E_;
    gather_f16<<<dim3((unsigned)((tot + 255) / 256)), blk, 0, stream>>>(
        enc_ids, emb, enc_emb16, tot, E_);
    tot = (long)B_ * TD_ * E_;
    gather_f16<<<dim3((unsigned)((tot + 255) / 256)), blk, 0, stream>>>(
        dec_ids, emb, dec_emb16, tot, E_);
  }

  // gx_enc = enc_emb @ enc_k + enc_bi   [8192,1536]
  gemm_wmma<<<dim3(G3_ / 256, (B_ * TE_) / 32), blk, 0, stream>>>(
      enc_emb16, enc_kT, enc_bi, gx_enc, nullptr, B_ * TE_, G3_, E_, E_);

  // gxe = dec_emb @ dec_k[H:H+E] + dec_bi   [2048,1536]
  gemm_wmma<<<dim3(G3_ / 256, (B_ * TD_) / 32), blk, 0, stream>>>(
      dec_emb16, dec_kT + H_, dec_bi, gxe, nullptr, B_ * TD_, G3_, E_, H_ + E_);

  // encoder recurrence
  enc_scan<<<1, 1024, 0, stream>>>(gx_enc, enc_rkT, enc_br, enc_ids, enc16, h_final);

  // keys = enc_out @ W2 + b2 (f16 out)   [8192,512]
  gemm_wmma<<<dim3(H_ / 256, (B_ * TE_) / 32), blk, 0, stream>>>(
      enc16, W2T, b2, nullptr, keys16, B_ * TE_, H_, H_, H_);

  // decoder recurrence + attention
  dec_scan<<<1, 1024, 0, stream>>>(gxe, dec_kT, dec_rkT, dec_br, W1T, b1,
                                   keys16, enc16, Va, bV, dec_ids, h_final, decout16);

  // out = dec_out @ Wd + bd   [2048,32000]
  gemm_wmma<<<dim3(V_ / 256, (B_ * TD_) / 32), blk, 0, stream>>>(
      decout16, WdT, bd, out, nullptr, B_ * TD_, V_, H_, H_);
}
